// SortAttn_5617817223314
// MI455X (gfx1250) — matrix operand order repaired and checked by reading, not compiled, not run on hardware
//
#include <hip/hip_runtime.h>
#include <hip/hip_bf16.h>

// ---------------------------------------------------------------------------
// SortAttn fused pipeline for gfx1250 (MI455X), wave32 + WMMA bf16.
//   ref: sorted = sort(x,axis=1); h=[x;sorted] @ W1 ; shuffle ; @W2 ; shuffle ;
//        relu(@W3) ; sigmoid(@W4) ; out = x * scores
// GEMMs run as v_wmma_f32_16x16x32_bf16 with f32 accumulate.
// Column-per-wave tiling: one B fragment feeds MT independent accumulator
// chains (less L2 traffic, more XDL overlap).
// ---------------------------------------------------------------------------

typedef __attribute__((ext_vector_type(16))) __bf16 v16bf;
typedef __attribute__((ext_vector_type(8)))  float  v8f;

union FragBF { v16bf bf; uint4 q[2]; };   // 32 bytes: one WMMA A/B fragment per lane

// float -> bf16 (round to nearest even), stored as raw u16
__device__ __forceinline__ unsigned short f2bfu(float f) {
    unsigned int u = __float_as_uint(f);
    u += 0x7FFFu + ((u >> 16) & 1u);
    return (unsigned short)(u >> 16);
}

// ---------------------------------------------------------------------------
// Weight conversion: fp32 (K,N) row-major  ->  bf16 in WMMA B-fragment order.
// Fragment (ks,nt): per lane 16 bf16:
//   lanes 0-15 : K = ks*32 + e        (e=0..15), N = nt*16 + lane
//   lanes16-31 : K = ks*32 + 16 + e,           N = nt*16 + (lane-16)
// Flat: out[(((ks*NT+nt)*32 + lane)*16 + e)]
// ---------------------------------------------------------------------------
__global__ void wconv_kernel(const float* __restrict__ W,
                             unsigned short* __restrict__ out, int K, int N) {
    int o = blockIdx.x * blockDim.x + threadIdx.x;
    if (o >= K * N) return;
    int NT     = N >> 4;
    int frag   = o >> 9;
    int within = o & 511;
    int lane   = within >> 4;
    int e      = within & 15;
    int ks     = frag / NT;
    int nt     = frag - ks * NT;
    int k      = ks * 32 + e + ((lane & 16) ? 16 : 0);
    int n      = nt * 16 + (lane & 15);
    out[o] = f2bfu(W[(size_t)k * N + n]);
}

// ---------------------------------------------------------------------------
// One output COLUMN (16 cols, MT*16 rows): for each 32-wide K step, load the
// B fragment once and feed MT independent accumulators.
// A-fragment (matches B fragment order): lane<16 -> K {0..15} of the
// 32-wide step; lane>=16 -> K {16..31}.
// ---------------------------------------------------------------------------
template <int MT, int NT, int KT>
__device__ __forceinline__ void wmma_col(const unsigned short* __restrict__ A, int lda,
                                         const unsigned short* __restrict__ Bf,
                                         int nt, int lane, v8f acc[MT]) {
    const unsigned short* abase = A + (size_t)(lane & 15) * lda + ((lane & 16) ? 16 : 0);
    const unsigned short* bbase = Bf + ((size_t)nt * 32 + lane) * 16;
    for (int ks = 0; ks < KT; ++ks) {
        FragBF b;
        const unsigned short* bp = bbase + (size_t)ks * NT * 512;
        b.q[0] = *(const uint4*)(bp);                      // global_load_b128
        b.q[1] = *(const uint4*)(bp + 8);                  // global_load_b128
#pragma unroll
        for (int mt = 0; mt < MT; ++mt) {
            FragBF a;
            const unsigned short* ap = abase + (size_t)mt * 16 * lda + ks * 32;
            a.q[0] = *(const uint4*)(ap);                  // ds_load_b128
            a.q[1] = *(const uint4*)(ap + 8);              // ds_load_b128
            acc[mt] = __builtin_amdgcn_wmma_f32_16x16x32_bf16(
                          false, a.bf, false, b.bf, (short)0, acc[mt], false, false);
        }
    }
}

// ---------------------------------------------------------------------------
// Fused main kernel: one workgroup (256 thr = 8 wave32) per 16 batch rows.
// Dynamic LDS: 3 x 32KB regions, ping-ponged across phases.
// ---------------------------------------------------------------------------
__global__ __launch_bounds__(256)
void sortattn_fused_kernel(const float* __restrict__ x,
                           const unsigned short* __restrict__ w1f,
                           const unsigned short* __restrict__ w2f,
                           const unsigned short* __restrict__ w3f,
                           const unsigned short* __restrict__ w4f,
                           float* __restrict__ out) {
    extern __shared__ unsigned char smem[];
    float*          r0f = (float*)(smem);                   // 32KB: x rows (fp32)
    float*          r1f = (float*)(smem + 32768);           // 32KB: sorted rows (fp32)
    unsigned short* r2h = (unsigned short*)(smem + 65536);  // 32KB: bf16 staging
    unsigned short* r0h = (unsigned short*)(smem);          // alias of r0f
    unsigned short* r1h = (unsigned short*)(smem + 32768);  // alias of r1f

    const int  tid  = threadIdx.x;
    const int  lane = tid & 31;
    const int  wave = tid >> 5;
    const long b0   = (long)blockIdx.x * 16;   // first batch row of this WG

    // ---- P1: load 16 rows of x into r0f, copy to r1f ------------------------
    {
        const float4* xg = (const float4*)(x + b0 * 512);
        float4* s0 = (float4*)r0f;
        float4* s1 = (float4*)r1f;
        for (int i = tid; i < 2048; i += 256) {   // 16*512/4
            float4 v = xg[i];
            s0[i] = v;
            s1[i] = v;
        }
    }
    __syncthreads();

    // ---- P1b: bitonic sort each 512-float row of r1f ascending --------------
    for (int k = 2; k <= 512; k <<= 1) {
        for (int j = k >> 1; j > 0; j >>= 1) {
            for (int pp = tid; pp < 16 * 256; pp += 256) {   // 256 compare pairs/row
                int row = pp >> 8;
                int p   = pp & 255;
                int i   = ((p & ~(j - 1)) << 1) | (p & (j - 1));
                int l   = i | j;
                float* base = r1f + row * 512;
                float a = base[i];
                float c = base[l];
                bool up = ((i & k) == 0);
                bool sw = up ? (a > c) : (a < c);
                if (sw) { base[i] = c; base[l] = a; }
            }
            __syncthreads();
        }
    }

    // ---- P2: build GEMM1 A matrix (32 x 512 bf16): row 2r=x_r, row 2r+1=sorted_r
    for (int i = tid; i < 32 * 512; i += 256) {
        int m = i >> 9, col = i & 511;
        const float* src = (m & 1) ? (r1f + (m >> 1) * 512) : (r0f + (m >> 1) * 512);
        r2h[i] = f2bfu(src[col]);
    }
    __syncthreads();

    // ---- P3: GEMM1 (32x512)@(512x512) + shuffle#1 scatter -> r0h ------------
    // per batch row: p = mm*512+n in h1-flat; shuffled index q = (p&255)*4 + p>>8
    for (int nt = wave; nt < 32; nt += 8) {
        v8f acc[2] = {};
        wmma_col<2, 32, 16>(r2h, 512, w1f, nt, lane, acc);
#pragma unroll
        for (int mt = 0; mt < 2; ++mt) {
#pragma unroll
            for (int r = 0; r < 8; ++r) {
                int m = mt * 16 + r + ((lane & 16) ? 8 : 0);
                int n = nt * 16 + (lane & 15);
                int p = ((m & 1) << 9) | n;
                int q = ((p & 255) << 2) | (p >> 8);
                r0h[(m >> 1) * 1024 + q] = f2bfu(acc[mt][r]);
            }
        }
    }
    __syncthreads();

    // ---- P4: GEMM2 (64x256)@(256x256) + shuffle#2 scatter -> r1h ------------
    // row m = b*4+g, col j: shuffled index q = j*4 + g
    for (int nt = wave; nt < 16; nt += 8) {
        v8f acc[4] = {};
        wmma_col<4, 16, 8>(r0h, 256, w2f, nt, lane, acc);
#pragma unroll
        for (int mt = 0; mt < 4; ++mt) {
#pragma unroll
            for (int r = 0; r < 8; ++r) {
                int m = mt * 16 + r + ((lane & 16) ? 8 : 0);
                int j = nt * 16 + (lane & 15);
                int q = (j << 2) | (m & 3);
                r1h[(m >> 2) * 1024 + q] = f2bfu(acc[mt][r]);
            }
        }
    }
    __syncthreads();

    // ---- P5: GEMM3 (64x256)@(256x256) + ReLU -> r2h (16 x 1024 plain) -------
    for (int nt = wave; nt < 16; nt += 8) {
        v8f acc[4] = {};
        wmma_col<4, 16, 8>(r1h, 256, w3f, nt, lane, acc);
#pragma unroll
        for (int mt = 0; mt < 4; ++mt) {
#pragma unroll
            for (int r = 0; r < 8; ++r) {
                int m = mt * 16 + r + ((lane & 16) ? 8 : 0);
                int j = nt * 16 + (lane & 15);
                float v = acc[mt][r] > 0.0f ? acc[mt][r] : 0.0f;
                r2h[m * 256 + j] = f2bfu(v);   // == b*1024 + g*256 + j
            }
        }
    }
    __syncthreads();

    // ---- P6: GEMM4 (16x1024)@(1024x512) + sigmoid + x*scores -> out ---------
    for (int nt = wave; nt < 32; nt += 8) {
        v8f acc[1] = {};
        wmma_col<1, 32, 32>(r2h, 1024, w4f, nt, lane, acc);
#pragma unroll
        for (int r = 0; r < 8; ++r) {
            int  m   = r + ((lane & 16) ? 8 : 0);
            int  col = nt * 16 + (lane & 15);
            long gi  = (b0 + m) * 512 + col;
            float s  = 1.0f / (1.0f + __expf(-acc[0][r]));
            out[gi]  = x[gi] * s;
        }
    }
}

// ---------------------------------------------------------------------------
extern "C" void kernel_launch(void* const* d_in, const int* in_sizes, int n_in,
                              void* d_out, int out_size, void* d_ws, size_t ws_size,
                              hipStream_t stream) {
    const float* x  = (const float*)d_in[0];   // (B, 512)
    const float* W1 = (const float*)d_in[1];   // (512, 512)
    const float* W2 = (const float*)d_in[2];   // (256, 256)
    const float* W3 = (const float*)d_in[3];   // (256, 256)
    const float* W4 = (const float*)d_in[4];   // (1024, 512)
    float* out = (float*)d_out;

    const int B = in_sizes[0] / 512;

    // bf16 fragment-swizzled weights in workspace (1.75 MB total -> L2 resident)
    unsigned short* w1f = (unsigned short*)d_ws;     // 512*512
    unsigned short* w2f = w1f + 512 * 512;           // 256*256
    unsigned short* w3f = w2f + 256 * 256;           // 256*256
    unsigned short* w4f = w3f + 256 * 256;           // 1024*512

    wconv_kernel<<<(512 * 512 + 255) / 256, 256, 0, stream>>>(W1, w1f, 512, 512);
    wconv_kernel<<<(256 * 256 + 255) / 256, 256, 0, stream>>>(W2, w2f, 256, 256);
    wconv_kernel<<<(256 * 256 + 255) / 256, 256, 0, stream>>>(W3, w3f, 256, 256);
    wconv_kernel<<<(1024 * 512 + 255) / 256, 256, 0, stream>>>(W4, w4f, 1024, 512);

    sortattn_fused_kernel<<<B / 16, 256, 98304, stream>>>(x, w1f, w2f, w3f, w4f, out);
}
